// PostProcess_15264313770374
// MI455X (gfx1250) — matrix operand order repaired and checked by reading, not compiled, not run on hardware
//
#include <hip/hip_runtime.h>
#include <math.h>

// DETR post-process for MI455X (gfx1250, wave32).
// Out layout (flat f32, reference return order):
//   [0,1200)            scores   [4,300]
//   [1200,2400)         labels   [4,300]  (stored as float)
//   [2400,7200)         boxes    [4,300,4]
//   [7200,7200+240.8M)  masks    [4,300,448,448] (fp16-rounded, stored f32)
//
// 28x28 -> 448x448 bilinear resize is separable: Out = W^T * M * W with
// W = 28x448 half-pixel-center weights (2 nonzeros per column), padded to
// K=32 so it maps onto V_WMMA_F32_16X16X32_F16. W fragments are synthesized
// analytically in registers (no LDS, no strided loads); the hot stage-2 loop
// is ds_load_b128 -> v_wmma -> streaming stores. The ~1 GB output write is
// the roofline (~41 us at 23.3 TB/s).

#define NSEL   300
#define OUT_HW 448
#define B_     4
#define Q_     900
#define C_     91
#define NFLAT  (Q_ * C_)   // 81900

typedef __attribute__((ext_vector_type(16))) _Float16 v16h;
typedef __attribute__((ext_vector_type(8)))  _Float16 v8h;
typedef __attribute__((ext_vector_type(8)))  float    v8f;

// ---------------- order-preserving float<->uint keys -----------------------
__device__ __forceinline__ unsigned orderKey(float x) {
    unsigned u = __float_as_uint(x);
    return (u & 0x80000000u) ? ~u : (u | 0x80000000u);
}
__device__ __forceinline__ float keyToFloat(unsigned k) {
    unsigned u = (k & 0x80000000u) ? (k & 0x7FFFFFFFu) : ~k;
    return __uint_as_float(u);
}

// ---------------- kernel 1: radix-select top-300 + boxes -------------------
__global__ __launch_bounds__(1024) void topk_kernel(
    const float* __restrict__ logits,   // [B, Q, C]
    const float* __restrict__ boxes,    // [B, Q, 4] cxcywh
    const int*   __restrict__ tsizes,   // [B, 2] (h, w)
    float* __restrict__ outScores,      // [B, NSEL]
    float* __restrict__ outLabels,      // [B, NSEL]
    float* __restrict__ outBoxes,       // [B, NSEL, 4]
    int*   __restrict__ qidx)           // [B, NSEL] selected query index
{
    __shared__ unsigned hist[256];
    __shared__ unsigned keys[512];
    __shared__ int      idxs[512];
    __shared__ int      sSelBin, sKAbove, sCntGt, sCntEq;

    const int b    = blockIdx.x;
    const int tid  = threadIdx.x;
    const int nthr = blockDim.x;
    const float* L = logits + (size_t)b * NFLAT;

    // 4-round radix select (8 bits/round) for the 300th-largest key.
    unsigned prefVal = 0, prefMask = 0;
    int kRemain = NSEL;
    for (int shift = 24; shift >= 0; shift -= 8) {
        for (int i = tid; i < 256; i += nthr) hist[i] = 0u;
        __syncthreads();
        for (int i = tid; i < NFLAT; i += nthr) {
            unsigned k = orderKey(L[i]);
            if ((k & prefMask) == prefVal)
                atomicAdd(&hist[(k >> shift) & 255u], 1u);
        }
        __syncthreads();
        if (tid == 0) {
            int acc = 0, sel = 0;
            for (int bin = 255; bin >= 0; --bin) {
                int c = (int)hist[bin];
                if (acc + c >= kRemain) { sel = bin; break; }
                acc += c;
            }
            sSelBin = sel; sKAbove = acc;
        }
        __syncthreads();
        prefVal  |= ((unsigned)sSelBin) << shift;
        prefMask |= 255u << shift;
        kRemain  -= sKAbove;
        __syncthreads();
    }
    const unsigned thresh = prefVal;   // key of the k-th largest element

    if (tid == 0) { sCntGt = 0; sCntEq = 0; }
    for (int i = tid; i < 512; i += nthr) { keys[i] = 0u; idxs[i] = 0; }
    __syncthreads();

    // strictly-greater elements first, then ties on the threshold key
    for (int i = tid; i < NFLAT; i += nthr) {
        unsigned k = orderKey(L[i]);
        if (k > thresh) {
            int p = atomicAdd(&sCntGt, 1);
            if (p < NSEL) { keys[p] = k; idxs[p] = i; }
        }
    }
    __syncthreads();
    const int m = sCntGt;
    for (int i = tid; i < NFLAT; i += nthr) {
        unsigned k = orderKey(L[i]);
        if (k == thresh) {
            int p = m + atomicAdd(&sCntEq, 1);
            if (p < NSEL) { keys[p] = k; idxs[p] = i; }
        }
    }
    __syncthreads();

    // bitonic sort, descending, over 512 (padding key = 0 sorts last)
    for (int kk = 2; kk <= 512; kk <<= 1) {
        for (int j = kk >> 1; j > 0; j >>= 1) {
            for (int i = tid; i < 512; i += nthr) {
                int ixj = i ^ j;
                if (ixj > i) {
                    unsigned ka = keys[i], kb = keys[ixj];
                    bool desc = ((i & kk) == 0);
                    bool sw   = desc ? (ka < kb) : (ka > kb);
                    if (sw) {
                        keys[i] = kb; keys[ixj] = ka;
                        int t = idxs[i]; idxs[i] = idxs[ixj]; idxs[ixj] = t;
                    }
                }
            }
            __syncthreads();
        }
    }

    const float img_h = (float)tsizes[b * 2 + 0];
    const float img_w = (float)tsizes[b * 2 + 1];
    for (int r = tid; r < NSEL; r += nthr) {
        float logit = keyToFloat(keys[r]);
        float score = 1.0f / (1.0f + __expf(-logit));
        int fi  = idxs[r];
        int q   = fi / C_;
        int lab = fi - q * C_;
        outScores[b * NSEL + r] = score;
        outLabels[b * NSEL + r] = (float)lab;
        const float* bx = boxes + ((size_t)(b * Q_ + q)) * 4;
        float cx = bx[0], cy = bx[1], w = bx[2], h = bx[3];
        float* ob = outBoxes + ((size_t)(b * NSEL + r)) * 4;
        ob[0] = (cx - 0.5f * w) * img_w;
        ob[1] = (cy - 0.5f * h) * img_h;
        ob[2] = (cx + 0.5f * w) * img_w;
        ob[3] = (cy + 0.5f * h) * img_h;
        qidx[b * NSEL + r] = q;
    }
}

// ---------------- analytic bilinear weight synthesis -----------------------
// Column j of W (28 -> 448, half-pixel centers, scale 1/16) has exactly two
// nonzeros: rows l0/l1 with weights w0/w1 (clamped edges collapse to w=1).
__device__ __forceinline__ void bilinW(int j, int& l0, int& l1,
                                       float& w0, float& w1) {
    float s  = (j + 0.5f) * 0.0625f - 0.5f;
    float fl = floorf(s);
    int   i0 = (int)fl;
    float f  = s - fl;
    int   i1 = i0 + 1;
    l0 = i0 < 0 ? 0 : (i0 > 27 ? 27 : i0);
    l1 = i1 < 0 ? 0 : (i1 > 27 ? 27 : i1);
    if (l0 == l1) { w0 = 1.0f; w1 = 0.0f; }
    else          { w0 = 1.0f - f; w1 = f; }
}

// B fragment of W (K=32 x N=16 tile): lane's column n, K split by lane half.
__device__ __forceinline__ v16h makeWfragB(int n, int hi) {
    int l0, l1; float w0, w1;
    bilinW(n, l0, l1, w0, w1);
    const int kb = hi * 8;
    v16h bfr;
    #pragma unroll
    for (int i = 0; i < 16; ++i) {
        int k = kb + i + (i < 8 ? 0 : 8);
        float w = (k == l0 ? w0 : 0.0f) + (k == l1 ? w1 : 0.0f);
        bfr[i] = (_Float16)w;
    }
    return bfr;
}

// A fragment of W^T (M=16 x K=32 tile `it`): lane's row = W column 16*it+ln.
__device__ __forceinline__ v16h makeWTfragA(int it, int ln, int hi) {
    int l0, l1; float w0, w1;
    bilinW(16 * it + ln, l0, l1, w0, w1);
    const int kb = hi * 16;      // hi lanes: K 16-31
    v16h a;
    #pragma unroll
    for (int i = 0; i < 8; ++i) {
        int k0 = kb + i, k1 = kb + 8 + i;
        a[i]     = (_Float16)((k0 == l0 ? w0 : 0.0f) + (k0 == l1 ? w1 : 0.0f));
        a[i + 8] = (_Float16)((k1 == l0 ? w0 : 0.0f) + (k1 == l1 ? w1 : 0.0f));
    }
    return a;
}

// ---------------- kernel 2: bilinear resize as WMMA GEMMs ------------------
// One block per selected mask (1200 blocks), 8 wave32s.
// Stage 1: T(32x448)   = M_pad(32x32) @ W(32x448)      -> 56 WMMAs
// Stage 2: Out(448x448)= W^T(448x32)  @ T(32x448)      -> 784 WMMAs
__global__ __launch_bounds__(256) void resize_kernel(
    const float* __restrict__ masks,    // [B, Q, 28, 28]
    const int*   __restrict__ qidx,     // [B*NSEL]
    float* __restrict__ outMasks)       // [B*NSEL, 448, 448]
{
    __shared__ _Float16 sA[32 * 32];          // padded fp16 source mask (row-major)
    __shared__ _Float16 sTB[28 * 32 * 16];    // T in B-fragment layout:
    // sTB[jt][lane][i] = T[(lane>>4)*16 + i][16*jt + (lane&15)]

    const int tid  = threadIdx.x;
    const int lane = tid & 31;
    const int wave = tid >> 5;
    const int ln   = lane & 15;
    const int hi   = lane >> 4;
    const int bk   = blockIdx.x;
    const int b    = bk / NSEL;
    const int q    = qidx[bk];

    // Load 28x28 mask, round to fp16 (matches astype(float16)), pad to 32x32.
    const float* src = masks + ((size_t)(b * Q_ + q)) * (28 * 28);
    for (int i = tid; i < 32 * 32; i += 256) {
        int r = i >> 5, c = i & 31;
        float v = (r < 28 && c < 28) ? src[r * 28 + c] : 0.0f;
        sA[i] = (_Float16)v;
    }
    __syncthreads();

    // ---- Stage 1: T = M_pad @ W, result written in B-fragment layout -------
    for (int t = wave; t < 2 * 28; t += 8) {       // wave-uniform loop
        const int mt = t & 1, nt = t >> 1;
        // A fragment: two contiguous 8-half runs -> 2x ds_load_b128
        const int arow = 16 * mt + ln;
        const v8h alo = *(const v8h*)&sA[arow * 32 + hi * 8];
        const v8h ahi = *(const v8h*)&sA[arow * 32 + 16 + hi * 8];
        v16h a = __builtin_shufflevector(alo, ahi,
                  0, 1, 2, 3, 4, 5, 6, 7, 8, 9, 10, 11, 12, 13, 14, 15);
        // B fragment of W: synthesized in registers (2 nonzeros per column)
        v16h bfr = makeWfragB(16 * nt + ln, hi);
        v8f c = {};
        c = __builtin_amdgcn_wmma_f32_16x16x32_f16(false, a, false, bfr,
                                                   (short)0, c, false, false);
        // D rows for this lane are contiguous i-indices of the B-fragment
        // layout -> pack to 8 halves, single ds_store_b128.
        v8h dh;
        #pragma unroll
        for (int r = 0; r < 8; ++r) dh[r] = (_Float16)c[r];
        // row = 16*mt + 8*hi + r  ->  jt=nt, lane'=mt*16+ln, i=8*hi+r
        *(v8h*)&sTB[((nt * 32 + mt * 16 + ln) * 16) + 8 * hi] = dh;
    }
    __syncthreads();

    // ---- Stage 2: Out = W^T @ T, stream to HBM -----------------------------
    float* dst = outMasks + (size_t)bk * (OUT_HW * OUT_HW);
    const int t0 = wave * 98, t1 = t0 + 98;        // 784 tiles / 8 waves
    int it = t0 / 28;
    int jt = t0 - it * 28;
    v16h a = makeWTfragA(it, ln, hi);
    for (int t = t0; t < t1; ++t) {
        // B fragment of T: 2x ds_load_b128 (contiguous per lane)
        const v8h blo = *(const v8h*)&sTB[(jt * 32 + lane) * 16];
        const v8h bhi = *(const v8h*)&sTB[(jt * 32 + lane) * 16 + 8];
        v16h bfr = __builtin_shufflevector(blo, bhi,
                    0, 1, 2, 3, 4, 5, 6, 7, 8, 9, 10, 11, 12, 13, 14, 15);
        v8f c = {};
        c = __builtin_amdgcn_wmma_f32_16x16x32_f16(false, a, false, bfr,
                                                   (short)0, c, false, false);
        const int n = 16 * jt + ln;
        #pragma unroll
        for (int r = 0; r < 8; ++r) {
            int row = 16 * it + r + 8 * hi;
            // emulate fp16 output precision, store as f32 per harness layout
            float v = (float)(_Float16)c[r];
            dst[(size_t)row * OUT_HW + n] = v;
        }
        // advance (wave-uniform; EXEC stays all-ones for WMMA)
        if (++jt == 28) {
            jt = 0;
            ++it;
            if (t + 1 < t1) a = makeWTfragA(it, ln, hi);
        }
    }
}

// ---------------------------------------------------------------------------
extern "C" void kernel_launch(void* const* d_in, const int* in_sizes, int n_in,
                              void* d_out, int out_size, void* d_ws, size_t ws_size,
                              hipStream_t stream) {
    const float* logits = (const float*)d_in[0];   // [4,900,91]
    const float* boxes  = (const float*)d_in[1];   // [4,900,4]
    const float* masks  = (const float*)d_in[2];   // [4,900,28,28]
    const int*   tsz    = (const int*)d_in[3];     // [4,2]

    float* out       = (float*)d_out;
    float* outScores = out;
    float* outLabels = out + B_ * NSEL;
    float* outBoxes  = out + B_ * NSEL * 2;
    float* outMasks  = out + B_ * NSEL * 2 + B_ * NSEL * 4;
    int*   qidx      = (int*)d_ws;                 // B_*NSEL ints of scratch

    topk_kernel<<<B_, 1024, 0, stream>>>(logits, boxes, tsz,
                                         outScores, outLabels, outBoxes, qidx);
    resize_kernel<<<B_ * NSEL, 256, 0, stream>>>(masks, qidx, outMasks);
}
